// ModulatedConv2d_14474039787656
// MI455X (gfx1250) — compile-verified, matrix operand order checked by reading, not run on hardware
//
#include <hip/hip_runtime.h>
#include <hip/hip_bf16.h>

typedef __attribute__((ext_vector_type(16))) _Float16     v16h;
typedef __attribute__((ext_vector_type(8)))  float        v8f;
typedef __attribute__((ext_vector_type(4)))  unsigned int v4u;
typedef __attribute__((ext_vector_type(8)))  int          v8i;
typedef __attribute__((ext_vector_type(4)))  int          v4i;

#define N_      8
#define CIN     512
#define COUT    512
#define HW      64
#define NPIX    (HW*HW)          // 4096
#define KTOT    (CIN*9)          // 4608
#define WDIM    512
#define LDSS    40               // f16 stride per LDS row (32 + 8 pad) = 80B

#define USE_TDM __has_builtin(__builtin_amdgcn_tensor_load_to_lds)

// ---------------------------------------------------------------------------
// LDS byte offset of a __shared__ object (generic -> AS3 addrspacecast)
// ---------------------------------------------------------------------------
typedef __attribute__((address_space(3))) void lds_void_t;
__device__ __forceinline__ unsigned int lds_offset(void* p) {
    return (unsigned int)(uintptr_t)(lds_void_t*)p;
}

#if USE_TDM
// ---------------------------------------------------------------------------
// TDM: load a [rows x 32] f16 tile (row stride KTOT f16 in global) into LDS
// with a 16B pad inserted after every 64B row -> LDS row stride 80B (= LDSS).
// D# bit packing per CDNA5 ISA §8 (group0 128b, group1 256b; groups 2/3 zero
// => 2D tensor). data_size code 1 = 2 bytes. pad_interval code 3 = 16 DWORDs,
// pad_amount code 3 = 4 DWORDs.
// ---------------------------------------------------------------------------
__device__ __forceinline__ void tdm_load_a_tile(unsigned long long gaddr,
                                                unsigned int lds_addr,
                                                int rows) {
    v4u g0;
    g0[0] = 1u;                                        // count=1, user mode
    g0[1] = lds_addr;                                  // lds_addr
    g0[2] = (unsigned int)(gaddr & 0xFFFFFFFFull);     // global_addr[31:0]
    g0[3] = (unsigned int)((gaddr >> 32) & 0x1FFFFFFull)
          | (2u << 30);                                // type=2 ("image")

    const unsigned int dim0 = KTOT;        // tensor_dim0 (elements)
    const unsigned int dim1 = (unsigned int)rows;
    const unsigned long long s0 = KTOT;    // tensor_dim0_stride (elements)

    v8i g1;
    g1[0] = (int)((1u << 16)       // data_size = 2B
                | (1u << 20)       // pad_enable
                | (3u << 22)       // pad_interval: 16 DWORDs (64B)
                | (3u << 25));     // pad_amount:   4 DWORDs (16B)
    g1[1] = (int)((dim0 & 0xFFFFu) << 16);                       // dim0[15:0]
    g1[2] = (int)((dim0 >> 16) | ((dim1 & 0xFFFFu) << 16));      // dim0[31:16] | dim1[15:0]
    g1[3] = (int)((dim1 >> 16) | (32u << 16));                   // dim1[31:16] | tile_dim0=32
    g1[4] = (int)(dim1 & 0xFFFFu);                               // tile_dim1=rows, tile_dim2=0
    g1[5] = (int)(unsigned int)(s0 & 0xFFFFFFFFull);             // dim0_stride[31:0]
    g1[6] = (int)(unsigned int)((s0 >> 32) & 0xFFFFull);         // dim0_stride[47:32]
    g1[7] = 0;

    const v4i gz = {0, 0, 0, 0};
#if __has_include(<hip/amd_detail/amd_gfx1250_TDM.h>)
    const v8i gz8 = {0, 0, 0, 0, 0, 0, 0, 0};
    __builtin_amdgcn_tensor_load_to_lds(g0, g1, gz, gz, gz8, 0);   // 6-arg toolchain
#else
    __builtin_amdgcn_tensor_load_to_lds(g0, g1, gz, gz, 0);        // 5-arg toolchain
#endif
}
#endif // USE_TDM

// ---------------------------------------------------------------------------
// 1) style[n][ci] = dot(w[n,:], style_w[ci,:]) + style_b[ci]
// ---------------------------------------------------------------------------
__global__ __launch_bounds__(256) void style_kernel(
    const float* __restrict__ w, const float* __restrict__ sw,
    const float* __restrict__ sb, float* __restrict__ style)
{
    int idx = blockIdx.x * 256 + threadIdx.x;      // 0..4095
    int n = idx >> 9, ci = idx & 511;
    const float* wr = w  + (size_t)n  * WDIM;
    const float* sr = sw + (size_t)ci * WDIM;
    float acc = sb[ci];
    for (int i = 0; i < WDIM; ++i) acc += wr[i] * sr[i];
    style[idx] = acc;
}

// ---------------------------------------------------------------------------
// 2) invd[n][co] = rsqrt( sum_ci style^2 * sum_k conv_w^2 + eps )
// ---------------------------------------------------------------------------
__global__ __launch_bounds__(256) void denom_kernel(
    const float* __restrict__ style, const float* __restrict__ cw,
    float* __restrict__ invd)
{
    int idx = blockIdx.x * 256 + threadIdx.x;      // 0..4095
    int n = idx >> 9, co = idx & 511;
    const float* st = style + (size_t)n * CIN;
    const float* wr = cw + (size_t)co * KTOT;
    float acc = 0.f;
    for (int ci = 0; ci < CIN; ++ci) {
        float s = st[ci];
        float ws = 0.f;
        #pragma unroll
        for (int k = 0; k < 9; ++k) { float x = wr[ci*9 + k]; ws += x * x; }
        acc += s * s * ws;
    }
    invd[idx] = rsqrtf(acc + 1e-6f);
}

// ---------------------------------------------------------------------------
// 3) wmod_f16[n][co][k] = conv_w[co][k] * style[n][ci(k)] * invd[n][co]
//    grid = (KTOT/256=18, COUT, N)
// ---------------------------------------------------------------------------
__global__ __launch_bounds__(256) void modw_kernel(
    const float* __restrict__ cw, const float* __restrict__ style,
    const float* __restrict__ invd, _Float16* __restrict__ wmod)
{
    int k  = blockIdx.x * 256 + threadIdx.x;       // 0..4607
    int co = blockIdx.y;
    int n  = blockIdx.z;
    int ci = k / 9;
    float v = cw[(size_t)co * KTOT + k] * style[n * CIN + ci] * invd[n * COUT + co];
    wmod[((size_t)(n * COUT + co)) * KTOT + k] = (_Float16)v;
}

// ---------------------------------------------------------------------------
// 4) implicit-GEMM conv, v_wmma_f32_16x16x32_f16 + TDM double-buffered A tile
//    grid = (HW/2=32 row-pairs, COUT/128=4, N=8), block = 256 (8 waves)
//    Workgroup tile: 128 Cout x 128 pixels (two image rows). K-step 32.
//    Wave tile: 32 Cout x 64 px  -> 8 WMMA per K-step per wave.
// ---------------------------------------------------------------------------
__global__ __launch_bounds__(256) void modconv_wmma_kernel(
    const float* __restrict__ image, const _Float16* __restrict__ wmod,
    float* __restrict__ out)
{
    const int y0     = blockIdx.x * 2;      // first of 2 pixel rows
    const int cobase = blockIdx.y * 128;
    const int n      = blockIdx.z;
    const int t      = threadIdx.x;
    const int lane   = t & 31;
    const int wave   = t >> 5;

    __shared__ __align__(16) _Float16 Asm[2][128 * LDSS];   // [buf][co_local][k]
    __shared__ __align__(16) _Float16 Bsm[2][128 * LDSS];   // [buf][px][k]  (B^T)

    const int mgrp = wave & 3;              // rows mgrp*32 .. +31
    const int ngrp = wave >> 2;             // px   ngrp*64 .. +63

    v8f acc[2][4] = {};                     // [A-frag][B-frag]

    const _Float16* wrow = wmod + (size_t)(n * COUT + cobase) * KTOT;
    const float*    img  = image + (size_t)n * CIN * NPIX;

    const int bpx = t & 127;                // B row: pixel 0..127 (2 img rows)
    const int bkg = t >> 7;                 // B k-half: 0 or 1 (16 k each)
    const int byy = bpx >> 6;               // image row within pair
    const int bxx = bpx & 63;               // x within row

#if USE_TDM
    const unsigned long long wrow_g = (unsigned long long)(uintptr_t)wrow;
    const unsigned int ldsA0 = lds_offset(&Asm[0][0]);
    const unsigned int ldsA1 = lds_offset(&Asm[1][0]);
    if (wave == 0) tdm_load_a_tile(wrow_g, ldsA0, 128);   // prologue: A(0)->buf0
#else
    const int aro = t >> 1;                 // fallback A row 0..127
    const int akc = (t & 1) * 16;           // k chunk 0 / 16
#endif

    const int iters = KTOT / 32;            // 144
    for (int it = 0; it < iters; ++it) {
        const int kbase = it * 32;
        const int buf   = it & 1;

        // ---- B tile: im2col gather, f32 -> f16, 16 values per thread ----
        int kidx = kbase + bkg * 16;
        int ci   = kidx / 9;
        int kk   = kidx - ci * 9;
        _Float16 bv[16];
        #pragma unroll
        for (int j = 0; j < 16; ++j) {
            int dy = kk / 3;
            int dx = kk - dy * 3;
            int yy = y0 + byy + dy - 1;
            int xx = bxx + dx - 1;
            float v = 0.f;
            if ((unsigned)yy < 64u && (unsigned)xx < 64u)
                v = img[((size_t)ci << 12) + (yy << 6) + xx];
            bv[j] = (_Float16)v;
            if (++kk == 9) { kk = 0; ++ci; }
        }

#if !USE_TDM
        uint4 av0 = *(const uint4*)(wrow + (size_t)aro * KTOT + kbase + akc);
#endif

        __syncthreads();   // all waves finished reading bufs of iteration it-1

        #pragma unroll
        for (int j = 0; j < 16; ++j) Bsm[buf][bpx * LDSS + bkg * 16 + j] = bv[j];

#if USE_TDM
        if (wave == 0) {
            if (it + 1 < iters) {
                tdm_load_a_tile(wrow_g + (unsigned long long)(kbase + 32) * 2,
                                buf ? ldsA0 : ldsA1, 128);
                // in-order TENSORcnt: <=1 outstanding => A(it) has landed
                __builtin_amdgcn_s_wait_tensorcnt(1);
            } else {
                // final step: no new issue; require full completion
                __builtin_amdgcn_s_wait_tensorcnt(0);
            }
        }
#else
        *(uint4*)(&Asm[buf][aro * LDSS + akc]) = av0;
#endif
        __syncthreads();   // B stored, A(it) resident

        // ---- fragments per ISA 16-bit layout ----
        // lanes 0-15: K 0..7 (V0-3), 16..23 (V4-7); lanes 16-31: +8
        const int half = (lane < 16) ? 0 : 8;
        const int r    = lane & 15;

        union frag_t { uint4 u[2]; v16h v; };
        frag_t af[2], bf[4];
        #pragma unroll
        for (int i = 0; i < 2; ++i) {
            const _Float16* ap = &Asm[buf][(mgrp * 32 + i * 16 + r) * LDSS];
            af[i].u[0] = *(const uint4*)(ap + half);
            af[i].u[1] = *(const uint4*)(ap + half + 16);
        }
        #pragma unroll
        for (int j = 0; j < 4; ++j) {
            const _Float16* bp = &Bsm[buf][(ngrp * 64 + j * 16 + r) * LDSS];
            bf[j].u[0] = *(const uint4*)(bp + half);
            bf[j].u[1] = *(const uint4*)(bp + half + 16);
        }

        #pragma unroll
        for (int i = 0; i < 2; ++i)
            #pragma unroll
            for (int j = 0; j < 4; ++j)
                acc[i][j] = __builtin_amdgcn_wmma_f32_16x16x32_f16(
                    false, af[i].v, false, bf[j].v, (short)0, acc[i][j],
                    false, false);
    }

    // ---- store: lanes 0-15 -> M=v, lanes 16-31 -> M=v+8; N = lane&15 ----
    // pixels of the two rows are contiguous in row-major memory
    const int mofs = (lane < 16) ? 0 : 8;
    const int col  = lane & 15;
    float* obase = out + (size_t)(n * COUT + cobase) * NPIX + y0 * HW;
    #pragma unroll
    for (int i = 0; i < 2; ++i) {
        #pragma unroll
        for (int j = 0; j < 4; ++j) {
            #pragma unroll
            for (int v = 0; v < 8; ++v) {
                int m = mgrp * 32 + i * 16 + mofs + v;
                obase[(size_t)m * NPIX + ngrp * 64 + j * 16 + col] = acc[i][j][v];
            }
        }
    }
}

// ---------------------------------------------------------------------------
extern "C" void kernel_launch(void* const* d_in, const int* in_sizes, int n_in,
                              void* d_out, int out_size, void* d_ws, size_t ws_size,
                              hipStream_t stream)
{
    const float* image = (const float*)d_in[0];   // (8,512,64,64)
    const float* w     = (const float*)d_in[1];   // (8,512)
    const float* cw    = (const float*)d_in[2];   // (512,512,3,3)
    const float* sw    = (const float*)d_in[3];   // (512,512)
    const float* sb    = (const float*)d_in[4];   // (512,)
    float* out = (float*)d_out;                   // (8,512,64,64)

    // workspace layout
    float*    style = (float*)d_ws;                       // 4096 f32
    float*    invd  = style + N_ * CIN;                   // 4096 f32
    _Float16* wmod  = (_Float16*)(invd + N_ * COUT);      // 8*512*4608 f16

    style_kernel<<<dim3((N_ * CIN) / 256), dim3(256), 0, stream>>>(w, sw, sb, style);
    denom_kernel<<<dim3((N_ * COUT) / 256), dim3(256), 0, stream>>>(style, cw, invd);
    modw_kernel<<<dim3(KTOT / 256, COUT, N_), dim3(256), 0, stream>>>(cw, style, invd, wmod);
    modconv_wmma_kernel<<<dim3(HW / 2, COUT / 128, N_), dim3(256), 0, stream>>>(image, wmod, out);
}
